// InstantNGPNeRF_58926951301170
// MI455X (gfx1250) — matrix operand order repaired and checked by reading, not compiled, not run on hardware
//
#include <hip/hip_runtime.h>

typedef _Float16 half_t;
typedef __attribute__((ext_vector_type(16))) _Float16 v16h;
typedef __attribute__((ext_vector_type(8)))  float    v8f;

#define LVLS   16
#define TSZ    (1u << 19)      // hash entries per level
#define NPB    128             // points per block (4 waves)
#define ASTR   72              // halves per activation row (64 + pad)

// floor(16 * (2^(1/3))^l) computed as numpy float64 would
__constant__ float kRes[LVLS] = {16.f, 20.f, 25.f, 32.f, 40.f, 50.f, 64.f, 80.f,
                                 101.f, 128.f, 161.f, 203.f, 256.f, 322.f, 406.f, 512.f};

// ---------------------------------------------------------------------------
// Prep: pack W0/W1/W2 (zero-padded) into WMMA B-fragment order (f16) + biases.
// Fragment f = ct*2 + kc holds B[k = kc*32 + (i>>3)*16 + (lane>>4)*8 + (i&7)][n = ct*16 + (lane&15)]
// at wf[layer_base + f*512 + lane*16 + i]  (contiguous 32B per lane).
// ---------------------------------------------------------------------------
__global__ void ngp_prep(const float* __restrict__ W0, const float* __restrict__ b0,
                         const float* __restrict__ W1, const float* __restrict__ b1,
                         const float* __restrict__ W2, const float* __restrict__ b2,
                         half_t* __restrict__ wf, float* __restrict__ bf) {
    const int tid = threadIdx.x;
    for (int e = tid; e < 9216; e += 256) {
        int layer, rem;
        if (e < 4096)      { layer = 0; rem = e; }
        else if (e < 8192) { layer = 1; rem = e - 4096; }
        else               { layer = 2; rem = e - 8192; }
        const int f  = rem >> 9;
        const int l  = (rem >> 4) & 31;
        const int i  = rem & 15;
        const int ct = f >> 1, kc = f & 1;
        const int n  = ct * 16 + (l & 15);
        const int k  = kc * 32 + ((i >> 3) * 16) + ((l >> 4) * 8) + (i & 7);
        float v = 0.0f;
        if (layer == 0)      { if (k < 35) v = W0[k * 64 + n]; }
        else if (layer == 1) { v = W1[k * 64 + n]; }
        else                 { if (n < 4)  v = W2[k * 4 + n]; }
        wf[e] = (half_t)v;
    }
    if (tid < 64) { bf[tid] = b0[tid]; bf[64 + tid] = b1[tid]; }
    if (tid < 16) { bf[128 + tid] = (tid < 4) ? b2[tid] : 0.0f; }
}

// ---------------------------------------------------------------------------
// Fragment helpers (ISA §7.12.2 layouts, wave32)
// ---------------------------------------------------------------------------
__device__ __forceinline__ v16h load_A(const half_t* act, int rbase, int kbase,
                                       int m16, int hi) {
    // A 16x32 f16: lane m (lo half-wave): halves 0..7 = K0..7, 8..15 = K16..23
    //              lane m+16:             halves 0..7 = K8..15, 8..15 = K24..31
    const half_t* r = act + (rbase + m16) * ASTR + kbase + hi * 8;
    v16h a;
#pragma unroll
    for (int i = 0; i < 8; ++i) a[i] = r[i];
#pragma unroll
    for (int i = 0; i < 8; ++i) a[8 + i] = r[16 + i];
    return a;
}

__device__ __forceinline__ v8f wmma32(v16h a, v16h b, v8f c) {
    return __builtin_amdgcn_wmma_f32_16x16x32_f16(false, a, false, b,
                                                  (short)0, c, false, false);
}

// One 128-row x 64-col dense layer with bias+ReLU, per-wave (32 rows).
__device__ __forceinline__ void dense64(const half_t* in, half_t* outp,
                                        const half_t* wfrag, const float* bias,
                                        int row0, int lane) {
    const int m16 = lane & 15, hi = lane >> 4;
#pragma unroll
    for (int rt = 0; rt < 2; ++rt) {
        const int rbase = row0 + rt * 16;
        v16h a0 = load_A(in, rbase, 0,  m16, hi);
        v16h a1 = load_A(in, rbase, 32, m16, hi);
#pragma unroll
        for (int ct = 0; ct < 4; ++ct) {
            v16h b0 = *(const v16h*)(wfrag + (ct * 2 + 0) * 512 + lane * 16);
            v16h b1 = *(const v16h*)(wfrag + (ct * 2 + 1) * 512 + lane * 16);
            v8f c = {};
            c = wmma32(a0, b0, c);
            c = wmma32(a1, b1, c);
            const int   n  = ct * 16 + m16;
            const float bn = bias[n];
#pragma unroll
            for (int rr = 0; rr < 8; ++rr) {
                float v = c[rr] + bn;
                v = fmaxf(v, 0.0f);
                outp[(rbase + rr + hi * 8) * ASTR + n] = (half_t)v;
            }
        }
    }
}

// ---------------------------------------------------------------------------
// Main fused kernel: hash-encode -> MLP(WMMA) -> sigmoid/relu head
// ---------------------------------------------------------------------------
__global__ __launch_bounds__(NPB) void ngp_main(
    const float* __restrict__ pos, const float* __restrict__ dirs,
    const float* __restrict__ table, const half_t* __restrict__ wf,
    const float* __restrict__ bf, float* __restrict__ out, int N) {
    __shared__ __align__(32) half_t act0[NPB * ASTR];
    __shared__ __align__(32) half_t act1[NPB * ASTR];
    __shared__ __align__(32) half_t wlds[9216];
    __shared__ float blds[144];

    const int tid = threadIdx.x;

    // stage weight fragments + biases into LDS
    {
        const uint4* s = (const uint4*)wf;
        uint4*       d = (uint4*)wlds;
        for (int e = tid; e < 9216 / 8; e += NPB) d[e] = s[e];
        for (int e = tid; e < 144; e += NPB) blds[e] = bf[e];
    }

    const int p  = blockIdx.x * NPB + tid;
    const int pc = p < N ? p : N - 1;

    // ---- hash-grid encoding: one point per thread ----
    {
        const float px = pos[pc * 3 + 0], py = pos[pc * 3 + 1], pz = pos[pc * 3 + 2];
        half_t* row = act0 + tid * ASTR;
        for (int lvl = 0; lvl < LVLS; ++lvl) {
            const float r = kRes[lvl];
            const float sx = px * r, sy = py * r, sz = pz * r;
            const float fx = floorf(sx), fy = floorf(sy), fz = floorf(sz);
            const float tx = sx - fx, ty = sy - fy, tz = sz - fz;
            const unsigned cx = (unsigned)fx, cy = (unsigned)fy, cz = (unsigned)fz;
            const unsigned hx0 = cx,                   hx1 = cx + 1u;
            const unsigned hy0 = cy * 2654435761u,     hy1 = (cy + 1u) * 2654435761u;
            const unsigned hz0 = cz * 805459861u,      hz1 = (cz + 1u) * 805459861u;
            const float2* tb = (const float2*)table + (size_t)lvl * TSZ;
            float a0 = 0.0f, a1 = 0.0f;
#pragma unroll
            for (int c = 0; c < 8; ++c) {
                const unsigned ox = (c >> 2) & 1u, oy = (c >> 1) & 1u, oz = c & 1u;
                const unsigned h  = (ox ? hx1 : hx0) ^ (oy ? hy1 : hy0) ^ (oz ? hz1 : hz0);
                const float2 ft = tb[h & (TSZ - 1u)];
                const float  w  = (ox ? tx : 1.0f - tx) * (oy ? ty : 1.0f - ty) *
                                  (oz ? tz : 1.0f - tz);
                a0 += w * ft.x;
                a1 += w * ft.y;
            }
            row[2 * lvl + 0] = (half_t)a0;
            row[2 * lvl + 1] = (half_t)a1;
        }
        const float dx = dirs[pc * 3 + 0], dy = dirs[pc * 3 + 1], dz = dirs[pc * 3 + 2];
        const float nrm = sqrtf(dx * dx + dy * dy + dz * dz);
        const float is  = 1.0f / fmaxf(nrm, 1e-12f);
        row[32] = (half_t)(dx * is);
        row[33] = (half_t)(dy * is);
        row[34] = (half_t)(dz * is);
#pragma unroll
        for (int c2 = 35; c2 < 64; ++c2) row[c2] = (half_t)0.0f;
    }
    __syncthreads();

    const int lane = tid & 31;
    const int row0 = (tid >> 5) * 32;   // each wave owns 32 rows
    const int m16  = lane & 15;
    const int hi   = lane >> 4;

    // layer 0: 64(padded from 35) -> 64, ReLU
    dense64(act0, act1, wlds, blds, row0, lane);
    __syncthreads();
    // layer 1: 64 -> 64, ReLU
    dense64(act1, act0, wlds + 4096, blds + 64, row0, lane);
    __syncthreads();

    // head: 64 -> 16(padded from 4); sigmoid(rgb) + relu(density) to global
    {
        const half_t* wf2 = wlds + 8192;
        const float*  b2l = blds + 128;
#pragma unroll
        for (int rt = 0; rt < 2; ++rt) {
            const int rbase = row0 + rt * 16;
            v16h a0 = load_A(act0, rbase, 0,  m16, hi);
            v16h a1 = load_A(act0, rbase, 32, m16, hi);
            v16h b0 = *(const v16h*)(wf2 + lane * 16);
            v16h b1 = *(const v16h*)(wf2 + 512 + lane * 16);
            v8f c = {};
            c = wmma32(a0, b0, c);
            c = wmma32(a1, b1, c);
            const int   n  = m16;
            const float bn = b2l[n];
            if (n < 4) {
#pragma unroll
                for (int rr = 0; rr < 8; ++rr) {
                    const int pgl = blockIdx.x * NPB + rbase + rr + hi * 8;
                    if (pgl < N) {
                        const float v = c[rr] + bn;
                        if (n < 3) out[pgl * 3 + n] = 1.0f / (1.0f + expf(-v));
                        else       out[(size_t)N * 3 + pgl] = fmaxf(v, 0.0f);
                    }
                }
            }
        }
    }
}

// ---------------------------------------------------------------------------
extern "C" void kernel_launch(void* const* d_in, const int* in_sizes, int n_in,
                              void* d_out, int out_size, void* d_ws, size_t ws_size,
                              hipStream_t stream) {
    const float* positions  = (const float*)d_in[0];
    const float* directions = (const float*)d_in[1];
    const float* hash_table = (const float*)d_in[2];
    const float* W0 = (const float*)d_in[3];
    const float* b0 = (const float*)d_in[4];
    const float* W1 = (const float*)d_in[5];
    const float* b1 = (const float*)d_in[6];
    const float* W2 = (const float*)d_in[7];
    const float* b2 = (const float*)d_in[8];
    float* out = (float*)d_out;

    const int N = in_sizes[0] / 3;

    half_t* wfrag = (half_t*)d_ws;                               // 9216 f16 = 18432 B
    float*  bf    = (float*)((char*)d_ws + 9216 * sizeof(half_t)); // 144 f32

    ngp_prep<<<dim3(1), dim3(256), 0, stream>>>(W0, b0, W1, b1, W2, b2, wfrag, bf);
    ngp_main<<<dim3((N + NPB - 1) / NPB), dim3(NPB), 0, stream>>>(
        positions, directions, hash_table, wfrag, bf, out, N);
}
